// CausalSelfAttention_83313775608159
// MI455X (gfx1250) — matrix operand order repaired and checked by reading, not compiled, not run on hardware
//
#include <hip/hip_runtime.h>
#include <hip/hip_bf16.h>

// ---------------------------------------------------------------------------
// CausalSelfAttention (RWKV-style time_weighting + post-softmax head mix)
// B=4, T=1024, C=1024, H=16, D=64  -- MI455X / gfx1250, wave32, WMMA f16
// GEMMs: TDM (tensor_load_to_lds) double-buffered weight staging in LDS.
// ---------------------------------------------------------------------------

typedef _Float16     half16 __attribute__((ext_vector_type(16)));
typedef _Float16     half8  __attribute__((ext_vector_type(8)));
typedef float        float8 __attribute__((ext_vector_type(8)));
typedef unsigned int uint4v __attribute__((ext_vector_type(4)));
typedef unsigned int uint8v __attribute__((ext_vector_type(8)));

union Half16U { half16 v; half8 h[2]; };

constexpr int kB = 4;
constexpr int kT = 1024;
constexpr int kC = 1024;
constexpr int kH = 16;
constexpr int kD = 64;

__device__ __forceinline__ float8 wmma_f16(half16 a, half16 b, float8 c) {
    return __builtin_amdgcn_wmma_f32_16x16x32_f16(false, a, false, b, (short)0, c,
                                                  false, false);
}

__device__ __forceinline__ float redmax16(float v) {
    #pragma unroll
    for (int off = 1; off < 16; off <<= 1) v = fmaxf(v, __shfl_xor(v, off, 32));
    return v;
}
__device__ __forceinline__ float redsum16(float v) {
    #pragma unroll
    for (int off = 1; off < 16; off <<= 1) v += __shfl_xor(v, off, 32);
    return v;
}

// --------------------------- prep kernels ----------------------------------

__global__ void cvt_f16_kernel(const float* __restrict__ src,
                               _Float16* __restrict__ dst, int n) {
    int i = blockIdx.x * blockDim.x + threadIdx.x;
    if (i < n) dst[i] = (_Float16)src[i];
}

// xs = concat(time_shift(x)[:, :, :C/2], x[:, :, C/2:]) as f16
__global__ void build_xs_kernel(const float* __restrict__ x,
                                _Float16* __restrict__ xs) {
    int i = blockIdx.x * blockDim.x + threadIdx.x;
    if (i >= kB * kT * kC) return;
    int c = i & (kC - 1);
    int t = (i >> 10) & (kT - 1);
    float v;
    if (c < kC / 2) v = (t > 0) ? x[i - kC] : 0.0f;
    else            v = x[i];
    xs[i] = (_Float16)v;
}

// --------------- WMMA GEMM with TDM weight staging: out = A * B^T ----------
// A:  [B*T, kC] f16 row-major (per-wave global b128 loads).
// Bw: [kC, kC] f16 row-major (row = out channel) -- staged into LDS by the
//     Tensor Data Mover, 64 rows x 32 K per chunk, double buffered.
// Workgroup = 4 waves: 4 M-tiles x one shared 64-col N block.
// mode: 0 -> f16 [B,H,T,D], 1 -> f16 [B,H,D,T] (vT), 2 -> f32 [B*T,kC].

__global__ __launch_bounds__(128)
void wmma_gemm_abT_tdm(const _Float16* __restrict__ A,
                       const _Float16* __restrict__ Bw,
                       const float* __restrict__ bias,
                       void* __restrict__ outp, int mode) {
    const int nt   = blockIdx.y;                 // 64-col block (0..15)
    const int wave = threadIdx.x >> 5;           // 0..3 -> M sub-tile
    const int lane = threadIdx.x & 31;
    const int lo = lane & 15, hi = lane >> 4;
    const int mt = blockIdx.x * 4 + wave;        // 16-row tile over B*T

    __shared__ __align__(16) _Float16 bbuf[2][64][32];   // 8 KB, double buf

    // A fragment: lane holds row (lane&15), K chunks at k0+hi*16 / +8
    const _Float16* arow = A + (size_t)(mt * 16 + lo) * kC + hi * 16;

    // --- TDM descriptor (uniform, lives in SGPRs) ---
    const uint64_t gbase =
        (uint64_t)(uintptr_t)(Bw + (size_t)(nt * 64) * kC);

    uint8v g1;
    g1[0] = 0x00010000u;                    // data_size = 2 bytes
    g1[1] = (1024u & 0xffffu) << 16;        // tensor_dim0 = 1024 (lo 16)
    g1[2] = (1024u >> 16) | ((1024u & 0xffffu) << 16);  // dim0 hi | dim1 lo
    g1[3] = (1024u >> 16) | (32u << 16);    // dim1 hi | tile_dim0 = 32
    g1[4] = 64u;                            // tile_dim1 = 64, tile_dim2 = 0
    g1[5] = 1024u;                          // tensor_dim0_stride = 1024
    g1[6] = 0u;
    g1[7] = 0u;

    auto tdm_issue = [&](int buf, int k0) {
        uint32_t lds = (uint32_t)(uintptr_t)(&bbuf[buf][0][0]);
        uint64_t ga  = gbase + (uint64_t)k0 * 2u;
        uint4v g0;
        g0[0] = 1u;                              // count=1, user descriptor
        g0[1] = lds;                             // lds_addr
        g0[2] = (uint32_t)ga;                    // global_addr[31:0]
        g0[3] = (uint32_t)(ga >> 32) | (2u << 30);  // addr[56:32] | type=2
        asm volatile("tensor_load_to_lds %0, %1" :: "s"(g0), "s"(g1) : "memory");
    };

    if (threadIdx.x == 0) {
        tdm_issue(0, 0);
        __builtin_amdgcn_s_wait_tensorcnt(0);
    }
    __syncthreads();

    float8 acc[4] = {};
    for (int kt = 0; kt < kC / 32; ++kt) {
        const int cur = kt & 1;
        if (threadIdx.x == 0 && kt + 1 < kC / 32)
            tdm_issue(1 - cur, (kt + 1) * 32);   // prefetch next chunk via TDM

        __builtin_prefetch((const void*)(arow + kt * 32 + 128), 0, 1);

        Half16U a;
        a.h[0] = *(const half8*)(arow + kt * 32);
        a.h[1] = *(const half8*)(arow + kt * 32 + 8);
        #pragma unroll
        for (int ns = 0; ns < 4; ++ns) {
            half16 bf = *(const half16*)&bbuf[cur][ns * 16 + lo][hi * 16];
            acc[ns] = wmma_f16(a.v, bf, acc[ns]);
        }

        if (threadIdx.x == 0 && kt + 1 < kC / 32)
            __builtin_amdgcn_s_wait_tensorcnt(0);
        __syncthreads();
    }

    _Float16* outh = (_Float16*)outp;
    float*    outf = (float*)outp;
    #pragma unroll
    for (int ns = 0; ns < 4; ++ns) {
        const int n = nt * 64 + ns * 16 + lo;
        const float bv = bias[n];
        #pragma unroll
        for (int r = 0; r < 8; ++r) {
            float v = acc[ns][r] + bv;
            int g = mt * 16 + r + hi * 8;      // global row over B*T
            int bb = g >> 10, t = g & (kT - 1);
            int h = n >> 6,  d = n & (kD - 1);
            if (mode == 0)
                outh[(((size_t)bb * kH + h) * kT + t) * kD + d] = (_Float16)v;
            else if (mode == 1)
                outh[(((size_t)bb * kH + h) * kD + d) * kT + t] = (_Float16)v;
            else
                outf[(size_t)g * kC + n] = v;
        }
    }
}

// -------------------- pass 1: softmax row stats (m, l) ---------------------
// grid.x = B*H, grid.y = T/16, 32 threads.  S = q k^T / 8 with causal mask.

__global__ __launch_bounds__(32)
void attn_stats_kernel(const _Float16* __restrict__ q,
                       const _Float16* __restrict__ k,
                       float* __restrict__ m_out,
                       float* __restrict__ l_out) {
    const int bh = blockIdx.x;
    const int t0 = blockIdx.y * 16;
    const int lane = threadIdx.x;
    const int lo = lane & 15, hi = lane >> 4;

    const _Float16* qb = q + ((size_t)bh * kT + t0) * kD;
    Half16U aq0, aq1;
    aq0.h[0] = *(const half8*)(qb + (size_t)lo * kD + hi * 16);
    aq0.h[1] = *(const half8*)(qb + (size_t)lo * kD + hi * 16 + 8);
    aq1.h[0] = *(const half8*)(qb + (size_t)lo * kD + 32 + hi * 16);
    aq1.h[1] = *(const half8*)(qb + (size_t)lo * kD + 32 + hi * 16 + 8);

    float rm[8], rl[8];
    #pragma unroll
    for (int r = 0; r < 8; ++r) { rm[r] = -__builtin_inff(); rl[r] = 0.0f; }

    for (int s0 = 0; s0 <= t0; s0 += 16) {
        const _Float16* krow = k + ((size_t)bh * kT + s0 + lo) * kD + hi * 16;
        half16 bk0 = *(const half16*)(krow);
        half16 bk1 = *(const half16*)(krow + 32);
        float8 s = {};
        s = wmma_f16(aq0.v, bk0, s);
        s = wmma_f16(aq1.v, bk1, s);
        #pragma unroll
        for (int r = 0; r < 8; ++r) {
            const int t = t0 + r + hi * 8;
            const int ss = s0 + lo;
            float sval = (ss <= t) ? s[r] * 0.125f : -__builtin_inff();
            float tmax = redmax16(sval);
            float newm = fmaxf(rm[r], tmax);
            float e = __expf(sval - newm);       // exp(-inf)=0 for masked
            float es = redsum16(e);
            rl[r] = rl[r] * __expf(rm[r] - newm) + es;
            rm[r] = newm;
        }
    }
    if (lo == 0) {
        #pragma unroll
        for (int r = 0; r < 8; ++r) {
            size_t idx = (size_t)bh * kT + t0 + r + hi * 8;
            m_out[idx] = rm[r];
            l_out[idx] = rl[r];
        }
    }
}

// ----------- pass 2: fused softmax * tw, head-mix, P @ V -------------------
// grid.x = B, grid.y = T/16, 512 threads = 16 waves.
// Phase 1: wave h builds p[h] (16x32 f16) in LDS.  Phase 2: wave o mixes
// A = sum_h W[o,h] p[h] and accumulates y[o] = A @ v[o] via WMMA.

__global__ __launch_bounds__(512)
void attn_fused_kernel(const _Float16* __restrict__ q,
                       const _Float16* __restrict__ k,
                       const _Float16* __restrict__ vT,
                       const float* __restrict__ tw,
                       const float* __restrict__ hmix,
                       const float* __restrict__ m_ws,
                       const float* __restrict__ l_ws,
                       _Float16* __restrict__ y) {
    const int b    = blockIdx.x;
    const int t0   = blockIdx.y * 16;
    const int wave = threadIdx.x >> 5;   // head h (phase 1) / out-head o (phase 2)
    const int lane = threadIdx.x & 31;
    const int lo = lane & 15, hi = lane >> 4;

    __shared__ __align__(32) _Float16 pbuf[16][16][32];  // [h][m][k] 16 KB

    const int bh = b * kH + wave;

    // q A-fragments for this wave's head (loop invariant)
    const _Float16* qb = q + ((size_t)bh * kT + t0) * kD;
    Half16U aq0, aq1;
    aq0.h[0] = *(const half8*)(qb + (size_t)lo * kD + hi * 16);
    aq0.h[1] = *(const half8*)(qb + (size_t)lo * kD + hi * 16 + 8);
    aq1.h[0] = *(const half8*)(qb + (size_t)lo * kD + 32 + hi * 16);
    aq1.h[1] = *(const half8*)(qb + (size_t)lo * kD + 32 + hi * 16 + 8);

    // softmax stats for this head's 16 rows (per lane: rows r + 8*hi)
    float mr[8], li[8];
    #pragma unroll
    for (int r = 0; r < 8; ++r) {
        size_t idx = (size_t)bh * kT + t0 + r + hi * 8;
        mr[r] = m_ws[idx];
        li[r] = 1.0f / l_ws[idx];
    }

    // head-mix row for this wave's output head
    float wm[16];
    #pragma unroll
    for (int i = 0; i < 16; ++i) wm[i] = hmix[wave * kH + i];

    // vT B-fragment row pointers (head = wave as output head)
    const _Float16* vrow[4];
    #pragma unroll
    for (int ns = 0; ns < 4; ++ns)
        vrow[ns] = vT + (((size_t)bh) * kD + ns * 16 + lo) * kT + hi * 16;

    float8 acc[4] = {};

    for (int s0 = 0; s0 < t0 + 16; s0 += 32) {
        // ---- phase 1: probabilities for head = wave into LDS ----
        #pragma unroll
        for (int sc = 0; sc < 2; ++sc) {
            const int s0c = s0 + sc * 16;
            const _Float16* krow =
                k + ((size_t)bh * kT + s0c + lo) * kD + hi * 16;
            half16 bk0 = *(const half16*)(krow);
            half16 bk1 = *(const half16*)(krow + 32);
            float8 s = {};
            s = wmma_f16(aq0.v, bk0, s);
            s = wmma_f16(aq1.v, bk1, s);
            #pragma unroll
            for (int r = 0; r < 8; ++r) {
                const int t  = t0 + r + hi * 8;
                const int ss = s0c + lo;
                float p = 0.0f;
                if (ss <= t) {
                    float val = s[r] * 0.125f;
                    p = __expf(val - mr[r]) * li[r] *
                        tw[((size_t)wave * kT + t) * kT + ss];
                }
                pbuf[wave][r + hi * 8][sc * 16 + lo] = (_Float16)p;
            }
        }
        __syncthreads();

        // ---- phase 2: mix heads, accumulate y[o] += Pmix @ v[o] ----
        float af[16];
        #pragma unroll
        for (int i = 0; i < 16; ++i) af[i] = 0.0f;
        #pragma unroll
        for (int h = 0; h < 16; ++h) {
            half8 p0 = *(const half8*)&pbuf[h][lo][hi * 16];
            half8 p1 = *(const half8*)&pbuf[h][lo][hi * 16 + 8];
            float w = wm[h];
            #pragma unroll
            for (int e = 0; e < 8; ++e) {
                af[e]     += w * (float)p0[e];
                af[8 + e] += w * (float)p1[e];
            }
        }
        half16 amix;
        #pragma unroll
        for (int i = 0; i < 16; ++i) amix[i] = (_Float16)af[i];

        #pragma unroll
        for (int ns = 0; ns < 4; ++ns) {
            half16 bv = *(const half16*)(vrow[ns] + s0);
            acc[ns] = wmma_f16(amix, bv, acc[ns]);
        }
        __syncthreads();
    }

    // store y as f16 [B, T, C] with head-interleaved channels
    #pragma unroll
    for (int ns = 0; ns < 4; ++ns)
        #pragma unroll
        for (int r = 0; r < 8; ++r)
            y[((size_t)b * kT + t0 + r + hi * 8) * kC + wave * 64 + ns * 16 + lo] =
                (_Float16)acc[ns][r];
}

// --------------------------------------------------------------------------

extern "C" void kernel_launch(void* const* d_in, const int* in_sizes, int n_in,
                              void* d_out, int out_size, void* d_ws, size_t ws_size,
                              hipStream_t stream) {
    const float* x    = (const float*)d_in[0];
    const float* Wq   = (const float*)d_in[1];
    const float* bq   = (const float*)d_in[2];
    const float* Wk   = (const float*)d_in[3];
    const float* bk   = (const float*)d_in[4];
    const float* Wv   = (const float*)d_in[5];
    const float* bv   = (const float*)d_in[6];
    const float* tw   = (const float*)d_in[7];
    const float* hmix = (const float*)d_in[8];
    const float* Wp   = (const float*)d_in[9];
    const float* bp   = (const float*)d_in[10];
    float* out = (float*)d_out;

    char* ws = (char*)d_ws;
    const size_t szBTC = (size_t)kB * kT * kC * sizeof(_Float16);  // 8 MB
    const size_t szW   = (size_t)kC * kC * sizeof(_Float16);       // 2 MB
    const size_t szML  = (size_t)kB * kH * kT * sizeof(float);     // 256 KB

    _Float16* xs_h = (_Float16*)ws;              ws += szBTC;
    _Float16* Wq_h = (_Float16*)ws;              ws += szW;
    _Float16* Wk_h = (_Float16*)ws;              ws += szW;
    _Float16* Wv_h = (_Float16*)ws;              ws += szW;
    _Float16* Wp_h = (_Float16*)ws;              ws += szW;
    _Float16* q_h  = (_Float16*)ws;              ws += szBTC;
    _Float16* k_h  = (_Float16*)ws;              ws += szBTC;
    _Float16* vT_h = (_Float16*)ws;              ws += szBTC;
    _Float16* y_h  = (_Float16*)ws;              ws += szBTC;
    float*    m_ws = (float*)ws;                 ws += szML;
    float*    l_ws = (float*)ws;                 ws += szML;

    // prep: weights -> f16, time-shifted xs -> f16
    const int nW = kC * kC;
    cvt_f16_kernel<<<(nW + 255) / 256, 256, 0, stream>>>(Wq, Wq_h, nW);
    cvt_f16_kernel<<<(nW + 255) / 256, 256, 0, stream>>>(Wk, Wk_h, nW);
    cvt_f16_kernel<<<(nW + 255) / 256, 256, 0, stream>>>(Wv, Wv_h, nW);
    cvt_f16_kernel<<<(nW + 255) / 256, 256, 0, stream>>>(Wp, Wp_h, nW);
    const int nX = kB * kT * kC;
    build_xs_kernel<<<(nX + 255) / 256, 256, 0, stream>>>(x, xs_h);

    // QKV projections (WMMA + TDM weight staging):
    //   q,k -> [B,H,T,D] f16,  v -> [B,H,D,T] f16
    dim3 ggrid(kB * kT / 16 / 4, kC / 64);
    wmma_gemm_abT_tdm<<<ggrid, 128, 0, stream>>>(xs_h, Wq_h, bq, (void*)q_h, 0);
    wmma_gemm_abT_tdm<<<ggrid, 128, 0, stream>>>(xs_h, Wk_h, bk, (void*)k_h, 0);
    wmma_gemm_abT_tdm<<<ggrid, 128, 0, stream>>>(xs_h, Wv_h, bv, (void*)vT_h, 1);

    // softmax row stats (exact two-pass; head-mix forbids online rescale)
    attn_stats_kernel<<<dim3(kB * kH, kT / 16), 32, 0, stream>>>(q_h, k_h, m_ws, l_ws);

    // fused: probs * time_weighting, head mix in LDS, P @ V  ->  y f16
    attn_fused_kernel<<<dim3(kB, kT / 16), 512, 0, stream>>>(
        q_h, k_h, vT_h, tw, hmix, m_ws, l_ws, y_h);

    // output projection -> f32 d_out
    wmma_gemm_abT_tdm<<<ggrid, 128, 0, stream>>>(y_h, Wp_h, bp, (void*)out, 2);
}
